// DualAttention_70231305225167
// MI455X (gfx1250) — matrix operand order repaired and checked by reading, not compiled, not run on hardware
//
#include <hip/hip_runtime.h>
#include <hip/hip_bf16.h>

// ---------------- types ----------------
typedef __bf16 v16bf __attribute__((ext_vector_type(16)));
typedef float  v8f   __attribute__((ext_vector_type(8)));
typedef unsigned int u32x4 __attribute__((ext_vector_type(4)));
typedef unsigned int u32x8 __attribute__((ext_vector_type(8)));

union FragB {            // one WMMA 16x16x32 bf16 operand (32B / lane)
    v16bf bf;
    u32x8 u;
    u32x4 q[2];
};

#define HW   4096
#define CCH  256
#define CR   64

// ---------------- bf16 helpers ----------------
__device__ __forceinline__ unsigned short f2bf(float f) {
    union { float f; unsigned u; } x; x.f = f;
    unsigned r = x.u + 0x7FFFu + ((x.u >> 16) & 1u);   // round-nearest-even
    return (unsigned short)(r >> 16);
}
__device__ __forceinline__ float bf2f(unsigned short h) {
    union { unsigned u; float f; } x; x.u = ((unsigned)h) << 16;
    return x.f;
}
__device__ __forceinline__ unsigned pack2bf(float a, float b) {
    return (unsigned)f2bf(a) | ((unsigned)f2bf(b) << 16);
}
__device__ __forceinline__ v8f wmma_bf16(const FragB& a, const FragB& b, v8f c) {
    return __builtin_amdgcn_wmma_f32_16x16x32_bf16(false, a.bf, false, b.bf,
                                                   (short)0, c, false, false);
}
__device__ __forceinline__ void zero8(v8f& v) {
    #pragma unroll
    for (int i = 0; i < 8; ++i) v[i] = 0.0f;
}

// ---------------- zero page for branch-free conv boundary handling ----------------
__global__ __launch_bounds__(256) void zero_page_kernel(unsigned short* __restrict__ zp) {
    zp[threadIdx.x] = 0;                                      // 256 ushorts = 512B
}

// ---------------- prep: x NCHW f32 -> NHWC bf16 ----------------
__global__ __launch_bounds__(256) void prep_x_kernel(const float* __restrict__ x,
                                                     unsigned short* __restrict__ xn) {
    size_t idx = (size_t)blockIdx.x * 256 + threadIdx.x;      // 4*256*4096
    int p = idx & (HW - 1);
    size_t bc = idx >> 12;
    int b = (int)(bc >> 8), c = (int)(bc & 255);
    xn[((size_t)b * HW + p) * CCH + c] = f2bf(x[idx]);
}

// ---------------- prep: conv weights [co][ci][3][3] f32 -> [t][co][ci] bf16 ----------------
__global__ __launch_bounds__(256) void pack_wconv_kernel(const float* __restrict__ w,
                                                         unsigned short* __restrict__ wp, int Co) {
    size_t idx = (size_t)blockIdx.x * 256 + threadIdx.x;
    size_t total = (size_t)Co * CCH * 9;
    if (idx >= total) return;
    int t = (int)(idx % 9);
    size_t rem = idx / 9;
    int ci = (int)(rem % CCH);
    int co = (int)(rem / CCH);
    wp[((size_t)t * Co + co) * CCH + ci] = f2bf(w[idx]);
}

// ---------------- prep: fuse weights f32 -> bf16 (already [co][512]) ----------------
__global__ __launch_bounds__(256) void pack_fw_kernel(const float* __restrict__ fw,
                                                      unsigned short* __restrict__ fwb) {
    size_t idx = (size_t)blockIdx.x * 256 + threadIdx.x;      // 256*512
    fwb[idx] = f2bf(fw[idx]);
}

// ---------------- SE: per-(b,c) spatial mean ----------------
__global__ __launch_bounds__(256) void se_mean_kernel(const float* __restrict__ x,
                                                      float* __restrict__ mean) {
    __shared__ float sm[256];
    int bc = blockIdx.x;                                      // b*256 + c
    const float* src = x + (size_t)bc * HW;
    float s = 0.0f;
    for (int i = threadIdx.x; i < HW; i += 256) s += src[i];
    sm[threadIdx.x] = s; __syncthreads();
    for (int off = 128; off > 0; off >>= 1) {
        if (threadIdx.x < (unsigned)off) sm[threadIdx.x] += sm[threadIdx.x + off];
        __syncthreads();
    }
    if (threadIdx.x == 0) mean[bc] = sm[0] * (1.0f / HW);
}

// ---------------- SE: fc1-relu-fc2-sigmoid ----------------
__global__ __launch_bounds__(256) void se_mlp_kernel(const float* __restrict__ mean,
                                                     const float* __restrict__ w1, const float* __restrict__ b1,
                                                     const float* __restrict__ w2, const float* __restrict__ b2,
                                                     float* __restrict__ scale) {
    int b = blockIdx.x;
    __shared__ float hid[16];
    const float* mb = mean + b * CCH;
    if (threadIdx.x < 16) {
        float s = b1[threadIdx.x];
        for (int c = 0; c < CCH; ++c) s += mb[c] * w1[threadIdx.x * CCH + c];
        hid[threadIdx.x] = fmaxf(s, 0.0f);
    }
    __syncthreads();
    int co = threadIdx.x;
    float s = b2[co];
    #pragma unroll
    for (int h = 0; h < 16; ++h) s += hid[h] * w2[co * 16 + h];
    scale[b * CCH + co] = 1.0f / (1.0f + __expf(-s));
}

// ---------------- SE apply: glob half of cat_t[b][p][256+c] ----------------
__global__ __launch_bounds__(256) void se_apply_kernel(const unsigned short* __restrict__ xn,
                                                       const float* __restrict__ scale,
                                                       unsigned short* __restrict__ cat_t) {
    size_t idx = (size_t)blockIdx.x * 256 + threadIdx.x;      // 4*4096*256, c fastest
    int c = idx & 255;
    size_t bp = idx >> 8;
    int b = (int)(bp >> 12);
    float v = bf2f(xn[idx]) * scale[b * CCH + c];
    cat_t[bp * 512 + 256 + c] = f2bf(v);
}

// ---------------- 3x3 conv via implicit-GEMM WMMA ----------------
// out(p,co) = sum_{t,ci} x_nhwc[b][p+off(t)][ci] * wp[t][co][ci]
// one wave per 16x16 output tile; K loop = 9 taps x 8 chunks of 32 channels.
// Boundary handling is branch-free: out-of-image taps read a zeroed page, so
// all im2col loads are unconditional and can be hoisted above the WMMAs.
__global__ __launch_bounds__(32) void conv3x3_kernel(const unsigned short* __restrict__ xn,
                                                     const unsigned short* __restrict__ wp,
                                                     const float* __restrict__ bias,
                                                     const unsigned short* __restrict__ zpage,
                                                     unsigned short* __restrict__ out,
                                                     int Co, int transposed) {
    int blk = blockIdx.x;
    int nct = Co >> 4;
    int ct = blk % nct; blk /= nct;
    int pt = blk & 255;
    int b  = blk >> 8;
    int lane = threadIdx.x, ln = lane & 15, lh = lane >> 4;

    int p = (pt << 4) + ln;                 // A-matrix row M = spatial pos
    int h = p >> 6, w = p & 63;

    // per-tap branch-free row pointers
    const unsigned short* xrows[9];
    #pragma unroll
    for (int t = 0; t < 9; ++t) {
        int hh = h + t / 3 - 1, ww = w + t % 3 - 1;
        bool valid = (hh >= 0) && (hh < 64) && (ww >= 0) && (ww < 64);
        xrows[t] = valid ? (xn + ((size_t)(b * HW + hh * 64 + ww)) * CCH) : zpage;
    }

    v8f acc; zero8(acc);

    #pragma unroll
    for (int t = 0; t < 9; ++t) {
        const unsigned short* xrow = xrows[t];
        const unsigned short* wrow = wp + ((size_t)t * Co + (ct << 4) + ln) * CCH;
        #pragma unroll
        for (int kc = 0; kc < 8; ++kc) {
            FragB a, bb;
            // A: lane=M(p), K = 16*lh + e (16 contiguous ci)
            a.q[0] = *(const u32x4*)(xrow + kc * 32 + lh * 16);
            a.q[1] = *(const u32x4*)(xrow + kc * 32 + lh * 16 + 8);
            // B: lane=N(co), K = 16*lh + e (16 contiguous ci)
            bb.q[0] = *(const u32x4*)(wrow + kc * 32 + lh * 16);
            bb.q[1] = *(const u32x4*)(wrow + kc * 32 + lh * 16 + 8);
            acc = wmma_bf16(a, bb, acc);
        }
    }

    int co = (ct << 4) + ln;                // C layout: lane holds N=co fixed
    float bsv = bias[co];
    if (transposed) {                       // q_t/k_t : [b][p][Co]
        #pragma unroll
        for (int r = 0; r < 8; ++r) {
            int pp = (pt << 4) + r + 8 * lh;
            out[((size_t)(b * HW + pp)) * Co + co] = f2bf(acc[r] + bsv);
        }
    } else {                                // v : [b][co][HW], p contiguous
        u32x4 wv;
        #pragma unroll
        for (int r2 = 0; r2 < 4; ++r2)
            wv[r2] = pack2bf(acc[2 * r2] + bsv, acc[2 * r2 + 1] + bsv);
        *(u32x4*)(out + ((size_t)(b * Co + co)) * HW + (pt << 4) + lh * 8) = wv;
    }
}

// ---------------- flash attention: one wave per (b, 16-query tile) ----------------
// S^T(j,i) = sum_c k_t[b][j][c] * q_t[b][i][c]   (Cr=64 -> 2 WMMA per j-tile)
// O(c,i)  += V(c,j) * P^T(j,i)  streamed over j in 32-steps with online softmax.
__global__ __launch_bounds__(32) void attn_kernel(const unsigned short* __restrict__ qt,
                                                  const unsigned short* __restrict__ kt,
                                                  const unsigned short* __restrict__ vm,
                                                  unsigned short* __restrict__ cat_t) {
    int blk = blockIdx.x;
    int b  = blk >> 8;
    int i0 = (blk & 255) << 4;
    int lane = threadIdx.x, ln = lane & 15, lh = lane >> 4;

    // Q as B-operand, preloaded once: lane=N(i), K=c contiguous
    FragB bq[2];
    const unsigned short* qrow = qt + ((size_t)(b * HW + i0 + ln)) * CR;
    #pragma unroll
    for (int kc = 0; kc < 2; ++kc) {
        bq[kc].q[0] = *(const u32x4*)(qrow + kc * 32 + lh * 16);
        bq[kc].q[1] = *(const u32x4*)(qrow + kc * 32 + lh * 16 + 8);
    }

    v8f acc[16];
    #pragma unroll
    for (int ct = 0; ct < 16; ++ct) zero8(acc[ct]);
    float m = -1e30f, l = 0.0f;
    const float sc = 0.125f;                 // 1/sqrt(Cr)

    for (int j0 = 0; j0 < HW; j0 += 32) {
        // hint-prefetch next j-block's K rows (global_prefetch_b8)
        if (j0 + 32 < HW) {
            __builtin_prefetch(kt + ((size_t)(b * HW + j0 + 32 + lane)) * CR, 0, 1);
        }
        // ---- two 16-j tiles of S^T ----
        v8f s[2];
        #pragma unroll
        for (int t = 0; t < 2; ++t) {
            zero8(s[t]);
            const unsigned short* krow = kt + ((size_t)(b * HW + j0 + t * 16 + ln)) * CR;
            #pragma unroll
            for (int kc = 0; kc < 2; ++kc) {
                FragB ak;                    // A: lane=M(j), K=c pattern
                ak.q[0] = *(const u32x4*)(krow + kc * 32 + lh * 16);
                ak.q[1] = *(const u32x4*)(krow + kc * 32 + lh * 16 + 8);
                s[t] = wmma_bf16(ak, bq[kc], s[t]);
            }
        }
        // ---- online softmax over columns i (lane&15); cross-half via shfl_xor(16) ----
        float mt = -1e30f;
        #pragma unroll
        for (int t = 0; t < 2; ++t)
            #pragma unroll
            for (int r = 0; r < 8; ++r) { float x = s[t][r] * sc; s[t][r] = x; mt = fmaxf(mt, x); }
        mt = fmaxf(mt, __shfl_xor(mt, 16, 32));
        float mnew = fmaxf(m, mt);
        float alpha = __expf(m - mnew);
        float ps = 0.0f;
        unsigned u0[4], u1[4];
        #pragma unroll
        for (int r2 = 0; r2 < 4; ++r2) {
            float p0a = __expf(s[0][2 * r2]     - mnew);
            float p0b = __expf(s[0][2 * r2 + 1] - mnew);
            float p1a = __expf(s[1][2 * r2]     - mnew);
            float p1b = __expf(s[1][2 * r2 + 1] - mnew);
            ps += p0a + p0b + p1a + p1b;
            u0[r2] = pack2bf(p0a, p0b);
            u1[r2] = pack2bf(p1a, p1b);
        }
        ps += __shfl_xor(ps, 16, 32);
        l = l * alpha + ps;
        m = mnew;
        // rescale O accumulators (per-lane alpha == per-column alpha)
        #pragma unroll
        for (int ct = 0; ct < 16; ++ct)
            #pragma unroll
            for (int r = 0; r < 8; ++r) acc[ct][r] *= alpha;

        // ---- assemble P^T B-operand for K=32 (tile0 -> lanes<16, tile1 -> lanes>=16) ----
        FragB pb;
        #pragma unroll
        for (int r2 = 0; r2 < 4; ++r2) {
            unsigned e0 = __shfl_xor(u0[r2], 16, 32);
            unsigned e1 = __shfl_xor(u1[r2], 16, 32);
            pb.u[r2]     = lh ? e1 : u0[r2];
            pb.u[r2 + 4] = lh ? u1[r2] : e0;
        }
        // ---- O(c,i) += V(c,j) * P^T, software-pipelined V fragment loads ----
        const unsigned short* vbase = vm + ((size_t)(b * CCH + ln)) * HW + j0;
        FragB av;
        av.q[0] = *(const u32x4*)(vbase + lh * 16);
        av.q[1] = *(const u32x4*)(vbase + lh * 16 + 8);
        #pragma unroll
        for (int ct = 0; ct < 16; ++ct) {
            FragB avn = av;
            if (ct < 15) {                   // issue next fragment's loads before WMMA
                const unsigned short* vrow = vbase + (size_t)(ct + 1) * 16 * HW;
                avn.q[0] = *(const u32x4*)(vrow + lh * 16);
                avn.q[1] = *(const u32x4*)(vrow + lh * 16 + 8);
            }
            acc[ct] = wmma_bf16(av, pb, acc[ct]);
            av = avn;
        }
    }

    // ---- epilogue: O/l  -> cat_t[b][i][c] (channels 0..255) ----
    float rl = 1.0f / l;
    unsigned short* dstbase = cat_t + ((size_t)(b * HW + i0 + ln)) * 512;
    #pragma unroll
    for (int ct = 0; ct < 16; ++ct) {
        u32x4 w;
        #pragma unroll
        for (int r2 = 0; r2 < 4; ++r2)
            w[r2] = pack2bf(acc[ct][2 * r2] * rl, acc[ct][2 * r2 + 1] * rl);
        *(u32x4*)(dstbase + ct * 16 + lh * 8) = w;
    }
}

// ---------------- fuse 1x1 conv: out(co,p) = fw[co][0:512] . cat_t[b][p][0:512] ----------------
__global__ __launch_bounds__(32) void fuse_kernel(const unsigned short* __restrict__ cat_t,
                                                  const unsigned short* __restrict__ fwb,
                                                  const float* __restrict__ fb,
                                                  float* __restrict__ out) {
    int blk = blockIdx.x;
    int ct = blk & 15;
    int pt = (blk >> 4) & 255;
    int b  = blk >> 12;
    int lane = threadIdx.x, ln = lane & 15, lh = lane >> 4;

    v8f acc; zero8(acc);
    const unsigned short* arow = fwb   + ((size_t)(ct * 16 + ln)) * 512;          // A: M=co
    const unsigned short* brow = cat_t + ((size_t)(b * HW + pt * 16 + ln)) * 512; // B: N=p

    FragB a, bb;                              // software-pipelined K loop
    a.q[0]  = *(const u32x4*)(arow + lh * 16);
    a.q[1]  = *(const u32x4*)(arow + lh * 16 + 8);
    bb.q[0] = *(const u32x4*)(brow + lh * 16);
    bb.q[1] = *(const u32x4*)(brow + lh * 16 + 8);
    #pragma unroll
    for (int kc = 0; kc < 16; ++kc) {
        FragB an = a, bn = bb;
        if (kc < 15) {
            an.q[0] = *(const u32x4*)(arow + (kc + 1) * 32 + lh * 16);
            an.q[1] = *(const u32x4*)(arow + (kc + 1) * 32 + lh * 16 + 8);
            bn.q[0] = *(const u32x4*)(brow + (kc + 1) * 32 + lh * 16);
            bn.q[1] = *(const u32x4*)(brow + (kc + 1) * 32 + lh * 16 + 8);
        }
        acc = wmma_bf16(a, bb, acc);
        a = an; bb = bn;
    }
    #pragma unroll
    for (int r = 0; r < 8; ++r) {
        int co = ct * 16 + r + 8 * lh;
        out[((size_t)(b * CCH + co)) * HW + pt * 16 + ln] = acc[r] + fb[co];
    }
}

// ---------------- host ----------------
extern "C" void kernel_launch(void* const* d_in, const int* in_sizes, int n_in,
                              void* d_out, int out_size, void* d_ws, size_t ws_size,
                              hipStream_t stream) {
    const float* x     = (const float*)d_in[0];
    const float* wq    = (const float*)d_in[1];
    const float* bq    = (const float*)d_in[2];
    const float* wk    = (const float*)d_in[3];
    const float* bk    = (const float*)d_in[4];
    const float* wv    = (const float*)d_in[5];
    const float* bv    = (const float*)d_in[6];
    const float* fc1_w = (const float*)d_in[7];
    const float* fc1_b = (const float*)d_in[8];
    const float* fc2_w = (const float*)d_in[9];
    const float* fc2_b = (const float*)d_in[10];
    const float* fuse_w= (const float*)d_in[11];
    const float* fuse_b= (const float*)d_in[12];
    float* out = (float*)d_out;

    char* ws = (char*)d_ws;
    auto carve = [&](size_t bytes) -> void* {
        void* p = (void*)ws;
        ws += (bytes + 255) & ~(size_t)255;
        return p;
    };
    unsigned short* xn    = (unsigned short*)carve(4ull * HW * CCH * 2);   // 8 MB  NHWC bf16
    unsigned short* wqp   = (unsigned short*)carve(9ull * 64  * CCH * 2);
    unsigned short* wkp   = (unsigned short*)carve(9ull * 64  * CCH * 2);
    unsigned short* wvp   = (unsigned short*)carve(9ull * CCH * CCH * 2);
    unsigned short* fwb   = (unsigned short*)carve((size_t)CCH * 512 * 2);
    unsigned short* qt    = (unsigned short*)carve(4ull * HW * CR * 2);    // [b][p][64]
    unsigned short* ktb   = (unsigned short*)carve(4ull * HW * CR * 2);    // [b][p][64]
    unsigned short* vvb   = (unsigned short*)carve(4ull * CCH * HW * 2);   // [b][c][p]
    unsigned short* catt  = (unsigned short*)carve(4ull * HW * 512 * 2);   // 16 MB [b][p][512]
    float* meanb  = (float*)carve(4 * CCH * sizeof(float));
    float* scaleb = (float*)carve(4 * CCH * sizeof(float));
    unsigned short* zpage = (unsigned short*)carve(256 * 2);               // zero page

    zero_page_kernel<<<1, 256, 0, stream>>>(zpage);
    prep_x_kernel<<<4 * HW * CCH / 256, 256, 0, stream>>>(x, xn);
    pack_wconv_kernel<<<(64 * CCH * 9 + 255) / 256, 256, 0, stream>>>(wq, wqp, 64);
    pack_wconv_kernel<<<(64 * CCH * 9 + 255) / 256, 256, 0, stream>>>(wk, wkp, 64);
    pack_wconv_kernel<<<(CCH * CCH * 9 + 255) / 256, 256, 0, stream>>>(wv, wvp, CCH);
    pack_fw_kernel<<<(CCH * 512) / 256, 256, 0, stream>>>(fuse_w, fwb);

    se_mean_kernel<<<4 * CCH, 256, 0, stream>>>(x, meanb);
    se_mlp_kernel<<<4, 256, 0, stream>>>(meanb, fc1_w, fc1_b, fc2_w, fc2_b, scaleb);
    se_apply_kernel<<<4 * HW * CCH / 256, 256, 0, stream>>>(xn, scaleb, catt);

    conv3x3_kernel<<<4 * 256 * 4,  32, 0, stream>>>(xn, wqp, bq, zpage, qt,  64, 1);
    conv3x3_kernel<<<4 * 256 * 4,  32, 0, stream>>>(xn, wkp, bk, zpage, ktb, 64, 1);
    conv3x3_kernel<<<4 * 256 * 16, 32, 0, stream>>>(xn, wvp, bv, zpage, vvb, CCH, 0);

    attn_kernel<<<4 * 256, 32, 0, stream>>>(qt, ktb, vvb, catt);

    fuse_kernel<<<4 * 256 * 16, 32, 0, stream>>>(catt, fwb, fuse_b, out);
}
